// ConvCaps_35536559407837
// MI455X (gfx1250) — matrix operand order repaired and checked by reading, not compiled, not run on hardware
//
#include <hip/hip_runtime.h>

typedef __attribute__((ext_vector_type(16))) _Float16 v16h;
typedef __attribute__((ext_vector_type(8)))  float    v8f;

#define Bd 8      // batch
#define Cd 33     // num caps (d)
#define Od 33     // out caps (o)
#define Kd 33     // contraction (l)
#define IH 96
#define IW 96
#define HW (IH*IW)

#define KPA 48    // padded K stride (halves) for A (Xc); bound check is compile-time
#define KPB 64    // padded K stride (halves) for B (Wt); all loads in-bounds
#define NP 48     // padded o dim (3 WMMA n-tiles)
#define NG 5      // distinct o1 groups per b

#define XC_BYTES (Bd*IH*KPA*2)            // 73728
#define WT_BYTES (NG*NP*KPB*2)            // 30720
#define UNION_BYTES (XC_BYTES + WT_BYTES) // 104448; reused for routing state

union HalfPack { uint32_t u; _Float16 h[2]; };
union FragU    { v16h v; uint32_t u[8]; };
union AccU     { v8f  v; float f[8]; };

__global__ __launch_bounds__(256) void convcaps_fused_kernel(
    const float* __restrict__ x,    // (8,33,96,96)
    const float* __restrict__ w,    // (33,1,33,33)
    float* __restrict__ out)        // (8,96,33,33)
{
    // u-tile, f16, layout [d][o][i] : 33*33*96 halves = 209088 B
    __shared__ _Float16 uT[Cd * Od * IH];
    // phase-overlapped region: GEMM staging first, routing state after
    __shared__ __align__(16) char smem[UNION_BYTES];

    const int tid  = threadIdx.x;
    const int lane = tid & 31;
    const int wave = tid >> 5;
    const int j = blockIdx.x;       // out_w position
    const int b = blockIdx.y;       // batch
    const int o1_0 = (b * Cd) >> 3; // first o1 group used by this b

    // ---------------- Phase 1: stage operands to LDS (f16) ----------------
    _Float16* Xc = (_Float16*)smem;               // [b'][i][l(KPA)]
    _Float16* Wt = (_Float16*)(smem + XC_BYTES);  // [g][o(NP)][l(KPB)]

    for (int idx = tid; idx < Bd * IH * KPA; idx += 256) {
        int bp  = idx / (IH * KPA);
        int rem = idx % (IH * KPA);
        int i   = rem / KPA;
        int l   = rem % KPA;
        float v = 0.f;
        if (l < Kd) v = x[(bp * Kd + l) * HW + i * IW + j];  // column gather
        Xc[idx] = (_Float16)v;
    }
    for (int idx = tid; idx < NG * NP * KPB; idx += 256) {
        int g   = idx / (NP * KPB);
        int rem = idx % (NP * KPB);
        int o   = rem / KPB;
        int l   = rem % KPB;
        float v = 0.f;
        if (o < Od && l < Kd) v = w[((o1_0 + g) * Od + o) * Kd + l];
        Wt[idx] = (_Float16)v;
    }
    __syncthreads();

    // ---------------- Phase 2: WMMA GEMM into uT ----------------
    // Per d: u[i][o] = Xc[b1] (96 x 33) @ Wt[g]^T (33 x 33), padded M=96,N=48,K=64
    const int hi  = (lane >= 16) ? 1 : 0;
    const int l15 = lane & 15;

    for (int t = wave; t < Cd * 6 * 3; t += 8) {      // wave-uniform: EXEC stays full
        int d  = t / 18;
        int r2 = t % 18;
        int mt = r2 / 3;       // i-tile (0..5)
        int nt = r2 % 3;       // o-tile (0..2)
        int bd = b * Cd + d;
        int b1 = bd & 7;
        int g  = (bd >> 3) - o1_0;

        const _Float16* Arow = Xc + (b1 * IH + mt * 16 + l15) * KPA; // A[m=i][k=l]
        const _Float16* Brow = Wt + (g * NP + nt * 16 + l15) * KPB;  // B[k=l][n=o] stored [n][k]

        v8f acc = {};
        #pragma unroll
        for (int ks = 0; ks < 2; ++ks) {
            FragU fa, fb;
            const int kbA = ks * 32 + (hi ? 8 : 0);
            const int kbB = ks * 32 + (hi ? 8 : 0);
            #pragma unroll
            for (int vv = 0; vv < 8; ++vv) {
                // ISA 16-bit A layout: v0..3 -> K=2v(+hi8), v4..7 -> K=16+2(v-4)(+hi8)
                // With KPA=48 the in-bounds test equals (vv<4 || ks==0) for BOTH
                // half-waves -> compile-time constant, no divergent guards.
                const bool inA = (vv < 4) || (ks == 0);
                int ka = kbA + ((vv < 4) ? (2 * vv) : (16 + 2 * (vv - 4)));
                fa.u[vv] = inA ? *(const uint32_t*)(Arow + ka) : 0u;
                // B layout matches A: v0..3 -> K=2v(+hi8), v4..7 -> K=16+2(v-4)(+hi8);
                // KPB=64 with zero padding -> always in-bounds.
                int kb = kbB + ((vv < 4) ? (2 * vv) : (16 + 2 * (vv - 4)));
                fb.u[vv] = *(const uint32_t*)(Brow + kb);
            }
            acc = __builtin_amdgcn_wmma_f32_16x16x32_f16(
                      false, fa.v, false, fb.v, (short)0, acc, false, false);
        }
        // D layout: VGPR r, lane L -> (m = mt*16 + r + hi*8, n = nt*16 + L%16)
        AccU au; au.v = acc;
        int n     = nt * 16 + l15;            // o
        int mbase = mt * 16 + (hi ? 8 : 0);   // i base (8 consecutive rows)
        if (n < Od) {
            uint32_t* dst = (uint32_t*)&uT[(d * Od + n) * IH + mbase];
            #pragma unroll
            for (int p = 0; p < 4; ++p) {
                HalfPack hp;
                hp.h[0] = (_Float16)au.f[2 * p];
                hp.h[1] = (_Float16)au.f[2 * p + 1];
                dst[p] = hp.u;
            }
        }
    }
    __syncthreads();   // uT done; staging region now reusable

    // ---------------- Phase 3: dynamic routing (LDS-resident) ----------------
    float* bl = (float*)smem;                       // [i][d] 12672 B
    float* cc = (float*)(smem + 12672);             // [i][d] 12672 B
    float* sv = (float*)(smem + 25344);             // [d][34] 4488 B (s, then v in place)
    float* md = (float*)(smem + 29832);             // [33]
    float* Sd = (float*)(smem + 29964);             // [33] (stores 1/sumexp)
    float* rs = (float*)(smem + 30096);             // [96] (stores 1/(rowsum+eps))

    for (int idx = tid; idx < IH * Cd; idx += 256) bl[idx] = 0.f;
    __syncthreads();

    for (int it = 0; it < 3; ++it) {
        // softmax over i (per d)
        if (tid < Cd) {
            const int d = tid;
            float m = -1e30f;
            for (int i = 0; i < IH; ++i) m = fmaxf(m, bl[i * Cd + d]);
            float s = 0.f;
            for (int i = 0; i < IH; ++i) s += __expf(bl[i * Cd + d] - m);
            md[d] = m; Sd[d] = 1.f / s;
        }
        __syncthreads();
        for (int idx = tid; idx < IH * Cd; idx += 256) {
            int d = idx % Cd;
            cc[idx] = __expf(bl[idx] - md[d]) * Sd[d];
        }
        __syncthreads();
        // normalize across d (per i), eps = 1e-10
        if (tid < IH) {
            float s = 0.f;
            for (int d = 0; d < Cd; ++d) s += cc[tid * Cd + d];
            rs[tid] = 1.f / (s + 1e-10f);
        }
        __syncthreads();
        for (int idx = tid; idx < IH * Cd; idx += 256) {
            cc[idx] *= rs[idx / Cd];
        }
        __syncthreads();

        // s[d][o] = sum_i cc[i][d] * uT[d][o][i]  (wave-level dots, i coalesced)
        for (int p = wave; p < Cd * Od; p += 8) {
            int d = p / Od, o = p % Od;
            const _Float16* up = &uT[(d * Od + o) * IH];
            float a = cc[(lane     ) * Cd + d] * (float)up[lane     ]
                    + cc[(lane + 32) * Cd + d] * (float)up[lane + 32]
                    + cc[(lane + 64) * Cd + d] * (float)up[lane + 64];
            #pragma unroll
            for (int off = 16; off >= 1; off >>= 1)
                a += __shfl_xor(a, off, 32);
            if (lane == 0) sv[d * 34 + o] = a;
        }
        __syncthreads();

        // squash over o (eps = 1e-8), v written in place over s
        if (tid < Cd) {
            const int d = tid;
            float sq = 0.f;
            for (int o = 0; o < Od; ++o) { float t2 = sv[d * 34 + o]; sq += t2 * t2; }
            float sc = (sq / (1.f + sq)) * rsqrtf(sq + 1e-8f);
            for (int o = 0; o < Od; ++o) sv[d * 34 + o] *= sc;
        }
        __syncthreads();

        if (it < 2) {
            // bl[i][d] += sum_o v[d][o] * uT[d][o][i]
            for (int idx = tid; idx < Cd * IH; idx += 256) {
                int d = idx / IH;
                int i = idx % IH;
                const _Float16* up = &uT[d * Od * IH + i];
                float a = 0.f;
                for (int o = 0; o < Od; ++o)
                    a += sv[d * 34 + o] * (float)up[o * IH];
                bl[i * Cd + d] += a;
            }
            __syncthreads();
        }
    }

    // out (B, out_w, C, O): coalesced store of v
    for (int idx = tid; idx < Cd * Od; idx += 256) {
        int d = idx / Od, o = idx % Od;
        out[((b * IW + j) * Cd + d) * Od + o] = sv[d * 34 + o];
    }
}

extern "C" void kernel_launch(void* const* d_in, const int* in_sizes, int n_in,
                              void* d_out, int out_size, void* d_ws, size_t ws_size,
                              hipStream_t stream) {
    (void)in_sizes; (void)n_in; (void)out_size; (void)d_ws; (void)ws_size;
    const float* x = (const float*)d_in[0];   // (8,33,96,96) fp32
    const float* w = (const float*)d_in[1];   // (33,1,33,33) fp32
    float* out = (float*)d_out;               // (8,96,33,33) fp32

    dim3 grid(IW, Bd);   // one workgroup per (j, b) column
    convcaps_fused_kernel<<<grid, 256, 0, stream>>>(x, w, out);
}
